// JointModel_80444737454387
// MI455X (gfx1250) — matrix operand order repaired and checked by reading, not compiled, not run on hardware
//
#include <hip/hip_runtime.h>
#include <math.h>

typedef _Float16 v16h __attribute__((ext_vector_type(16)));
typedef _Float16 v8h  __attribute__((ext_vector_type(8)));
typedef float    v8f  __attribute__((ext_vector_type(8)));

#define KSEG 30
#define H    128
#define LDA  136                     // padded row stride in halves (16B aligned)
#define WROWS   (H + 16)             // 128 rows W2^T + 16 rows W3^T(padded)
#define WHALVES (WROWS * LDA)        // 19584 halves = 39168 bytes
#define WCHUNKS (WHALVES / 8)        // 2448 x 16-byte chunks

// ---------- helpers ----------
__device__ __forceinline__ unsigned encF(float f) {
    unsigned u = __float_as_uint(f);
    return (u & 0x80000000u) ? ~u : (u | 0x80000000u);
}
__device__ __forceinline__ float decF(unsigned u) {
    unsigned b = (u & 0x80000000u) ? (u & 0x7FFFFFFFu) : ~u;
    return __uint_as_float(b);
}
__device__ __forceinline__ v16h pack16(const _Float16* pLo, const _Float16* pHi) {
    v8h lo = *(const v8h*)pLo;
    v8h hi = *(const v8h*)pHi;
    v16h r;
#pragma unroll
    for (int j = 0; j < 8; ++j) { r[j] = lo[j]; r[j + 8] = hi[j]; }
    return r;
}

// ---------- kernel 0: one-time weight pre-convert to f16 padded panels ----------
// wt rows [0,128): W2^T  (wt[n*LDA+k] = W2[k][n]);  rows [128,144): W3^T padded to 16 cols
__global__ void convert_kernel(const float* __restrict__ W2, const float* __restrict__ W3,
                               _Float16* __restrict__ wt) {
    const int t = threadIdx.x;
    for (int i = t; i < H * LDA; i += 256) {
        int n = i / LDA, k = i % LDA;
        wt[i] = (k < H) ? (_Float16)W2[k * H + n] : (_Float16)0.0f;
    }
    for (int i = t; i < 16 * LDA; i += 256) {
        int n = i / LDA, k = i % LDA;
        wt[H * LDA + i] = (n < 4 && k < H) ? (_Float16)W3[k * 4 + n] : (_Float16)0.0f;
    }
}

// ---------- kernel 1: init segment min/max encodings ----------
__global__ void seg_init_kernel(unsigned* maxE, unsigned* minE) {
    int t = threadIdx.x;
    if (t < KSEG * 3) { maxE[t] = 0u; minE[t] = 0xFFFFFFFFu; }
}

// ---------- kernel 2: fused MLP (WMMA layers 2 and 3), writes t and yaw ----------
__global__ void __launch_bounds__(128)
mlp_kernel(const float* __restrict__ pc1,
           const float* __restrict__ W1, const float* __restrict__ b1,
           const float* __restrict__ b2, const float* __restrict__ b3,
           const _Float16* __restrict__ wt,
           float* __restrict__ out, int N) {
    __shared__ __align__(16) _Float16 wts[WHALVES];   // W2^T + W3^T panels (f16)
    __shared__ __align__(16) _Float16 h1s[64 * LDA];  // h1 tile, then h2 tile
    __shared__ float pts[64 * 3];
    __shared__ float W1s[3 * H];
    __shared__ float b1s[H], b2s[H];

    const int t    = threadIdx.x;
    const int lane = t & 31;
    const int wave = t >> 5;
    const int m0   = blockIdx.x * 64;
    const int mCnt = min(64, N - m0);

    // ---- async-copy pre-converted weight panels straight into LDS (overlaps below) ----
    {
        const unsigned lbase = (unsigned)(uintptr_t)&wts[0];
        const unsigned long long gbase = (unsigned long long)(const void*)wt;
        for (int i = t; i < WCHUNKS; i += 128) {
            unsigned laddr = lbase + (unsigned)i * 16u;
            unsigned long long gaddr = gbase + (unsigned long long)i * 16ull;
            asm volatile("global_load_async_to_lds_b128 %0, %1, off"
                         :: "v"(laddr), "v"(gaddr) : "memory");
        }
    }

    // ---- stage small params + point tile while the async copy is in flight ----
    for (int i = t; i < 192; i += 128)
        pts[i] = (i < mCnt * 3) ? pc1[m0 * 3 + i] : 0.0f;
    for (int i = t; i < 3 * H; i += 128) W1s[i] = W1[i];
    if (t < H) { b1s[t] = b1[t]; b2s[t] = b2[t]; }
    __syncthreads();

    // ---- layer 1 (VALU, K=3, fp32 points): thread t owns column n = t ----
    {
        const float wa = W1s[t], wb = W1s[H + t], wc = W1s[2 * H + t], bb = b1s[t];
#pragma unroll 4
        for (int m = 0; m < 64; ++m) {
            float v = fmaf(pts[3 * m], wa, fmaf(pts[3 * m + 1], wb,
                      fmaf(pts[3 * m + 2], wc, bb)));
            h1s[m * LDA + t] = (_Float16)fmaxf(v, 0.0f);
        }
    }
    // all of this wave's async weight chunks must have landed; then block-wide barrier
    asm volatile("s_wait_asynccnt 0x0" ::: "memory");
    __syncthreads();

    // ---- layer 2: h2 = relu(h1 @ W2 + b2) via v_wmma_f32_16x16x32_f16 ----
    const int row  = lane & 15;
    const int half = lane >> 4;
    v8f acc[4][2];
#pragma unroll
    for (int tn = 0; tn < 2; ++tn) {
        float bias = b2s[(2 * wave + tn) * 16 + row];
#pragma unroll
        for (int tm = 0; tm < 4; ++tm)
#pragma unroll
            for (int r = 0; r < 8; ++r) acc[tm][tn][r] = bias;
    }
#pragma unroll
    for (int kk = 0; kk < 4; ++kk) {
        const int kb = kk * 32;
        v16h aF[4];
#pragma unroll
        for (int tm = 0; tm < 4; ++tm) {
            const _Float16* base = &h1s[(tm * 16 + row) * LDA + kb + half * 8];
            aF[tm] = pack16(base, base + 16);
        }
        v16h bF[2];
#pragma unroll
        for (int tn = 0; tn < 2; ++tn) {
            const _Float16* base = &wts[((2 * wave + tn) * 16 + row) * LDA + kb + half * 8];
            bF[tn] = pack16(base, base + 16);
        }
#pragma unroll
        for (int tm = 0; tm < 4; ++tm)
#pragma unroll
            for (int tn = 0; tn < 2; ++tn)
                acc[tm][tn] = __builtin_amdgcn_wmma_f32_16x16x32_f16(
                    false, aF[tm], false, bF[tn], (short)0, acc[tm][tn], false, false);
    }
    __syncthreads();   // all WMMA reads of h1s done -> safe to overwrite with h2

    // ---- relu + write h2 (f16) into h1s space (C-layout scatter) ----
#pragma unroll
    for (int tm = 0; tm < 4; ++tm)
#pragma unroll
        for (int tn = 0; tn < 2; ++tn) {
            const int n = (2 * wave + tn) * 16 + row;
#pragma unroll
            for (int r = 0; r < 8; ++r) {
                const int m = tm * 16 + r + (half ? 8 : 0);
                h1s[m * LDA + n] = (_Float16)fmaxf(acc[tm][tn][r], 0.0f);
            }
        }
    __syncthreads();

    // ---- layer 3: out = h2 @ W3pad + b3 via WMMA; wave owns M-tile = wave ----
    {
        v8f acc3;
        const float bias = (row < 4) ? b3[row] : 0.0f;
#pragma unroll
        for (int r = 0; r < 8; ++r) acc3[r] = bias;
#pragma unroll
        for (int kk = 0; kk < 4; ++kk) {
            const int kb = kk * 32;
            const _Float16* ab = &h1s[(wave * 16 + row) * LDA + kb + half * 8];
            v16h aF = pack16(ab, ab + 16);
            const _Float16* bb = &wts[(H + row) * LDA + kb + half * 8];
            v16h bF = pack16(bb, bb + 16);
            acc3 = __builtin_amdgcn_wmma_f32_16x16x32_f16(
                false, aF, false, bF, (short)0, acc3, false, false);
        }
        if (row < 4) {
            float* tOut   = out + (size_t)33 * N;
            float* yawOut = out + (size_t)36 * N;
#pragma unroll
            for (int r = 0; r < 8; ++r) {
                const int m  = wave * 16 + r + (half ? 8 : 0);
                const int gm = m0 + m;
                if (gm < N) {
                    if (row < 3) tOut[gm * 3 + row] = acc3[r];
                    else         yawOut[gm]         = acc3[r];
                }
            }
        }
    }
}

// ---------- kernel 3: softmax + argmax + segment min/max atomics ----------
__global__ void __launch_bounds__(256)
softmax_kernel(const float* __restrict__ logits, const float* __restrict__ pc1,
               float* __restrict__ out, int* __restrict__ ind,
               unsigned* __restrict__ gMax, unsigned* __restrict__ gMin, int N) {
    __shared__ unsigned lMax[KSEG * 3], lMin[KSEG * 3];
    const int t = threadIdx.x;
    if (t < KSEG * 3) { lMax[t] = 0u; lMin[t] = 0xFFFFFFFFu; }
    __syncthreads();

    const int i = blockIdx.x * 256 + t;
    if (i < N) {
        const float* row = logits + (size_t)i * KSEG;
        float x[KSEG];
        float mx = -INFINITY; int am = 0;
#pragma unroll
        for (int j = 0; j < KSEG; ++j) {
            x[j] = row[j];
            if (x[j] > mx) { mx = x[j]; am = j; }
        }
        float s = 0.0f;
#pragma unroll
        for (int j = 0; j < KSEG; ++j) { x[j] = __expf(x[j] - mx); s += x[j]; }
        const float inv = 1.0f / s;
        float* mrow = out + (size_t)3 * N + (size_t)i * KSEG;
#pragma unroll
        for (int j = 0; j < KSEG; ++j) mrow[j] = x[j] * inv;
        ind[i] = am;
#pragma unroll
        for (int c = 0; c < 3; ++c) {
            unsigned e = encF(pc1[i * 3 + c]);
            atomicMax(&lMax[am * 3 + c], e);
            atomicMin(&lMin[am * 3 + c], e);
        }
    }
    __syncthreads();
    if (t < KSEG * 3) {
        atomicMax(&gMax[t], lMax[t]);
        atomicMin(&gMin[t], lMin[t]);
    }
}

// ---------- kernel 4: segment centers ----------
__global__ void centers_kernel(const unsigned* maxE, const unsigned* minE, float* xc) {
    int t = threadIdx.x;
    if (t < KSEG * 3) xc[t] = 0.5f * (decF(maxE[t]) + decF(minE[t]));
}

// ---------- kernel 5: rotation + flow ----------
__global__ void __launch_bounds__(256)
flow_kernel(const float* __restrict__ pc1, const int* __restrict__ ind,
            const float* __restrict__ xc, float* __restrict__ out, int N) {
    __shared__ float xcs[KSEG * 3];
    const int t = threadIdx.x;
    if (t < KSEG * 3) xcs[t] = xc[t];
    __syncthreads();

    const int i = blockIdx.x * 256 + t;
    if (i >= N) return;
    const float px = pc1[i * 3 + 0], py = pc1[i * 3 + 1], pz = pc1[i * 3 + 2];
    const int k = ind[i];
    const float cx = xcs[k * 3 + 0], cy = xcs[k * 3 + 1], cz = xcs[k * 3 + 2];
    const float* tOut = out + (size_t)33 * N;
    const float tx = tOut[i * 3 + 0], ty = tOut[i * 3 + 1], tz = tOut[i * 3 + 2];
    const float yaw = out[(size_t)36 * N + i];
    float cs, sn;
    __sincosf(yaw, &sn, &cs);
    const float dx = px - cx, dy = py - cy, dz = pz - cz;
    const float rx = cs * dx - sn * dy;
    const float ry = sn * dx + cs * dy;
    out[i * 3 + 0] = rx + cx + tx - px;
    out[i * 3 + 1] = ry + cy + ty - py;
    out[i * 3 + 2] = dz + cz + tz - pz;   // rz == dz
}

extern "C" void kernel_launch(void* const* d_in, const int* in_sizes, int n_in,
                              void* d_out, int out_size, void* d_ws, size_t ws_size,
                              hipStream_t stream) {
    const float* pc1    = (const float*)d_in[0];
    const float* logits = (const float*)d_in[1];
    const float* W1 = (const float*)d_in[2];
    const float* b1 = (const float*)d_in[3];
    const float* W2 = (const float*)d_in[4];
    const float* b2 = (const float*)d_in[5];
    const float* W3 = (const float*)d_in[6];
    const float* b3 = (const float*)d_in[7];
    float* out = (float*)d_out;
    const int N = in_sizes[0] / 3;

    // workspace layout (u32 units): [0,96) maxEnc  [96,192) minEnc  [192,288) xc
    // [288, 288+N) ind  then (16B-aligned) f16 weight panels
    unsigned* maxE = (unsigned*)d_ws;
    unsigned* minE = maxE + 96;
    float*    xc   = (float*)(maxE + 192);
    int*      ind  = (int*)(maxE + 288);
    unsigned  wtOff = ((288u + (unsigned)N) + 3u) & ~3u;   // u32 units, 16B aligned
    _Float16* wt   = (_Float16*)(maxE + wtOff);

    convert_kernel<<<1, 256, 0, stream>>>(W2, W3, wt);
    seg_init_kernel<<<1, 128, 0, stream>>>(maxE, minE);
    mlp_kernel<<<(N + 63) / 64, 128, 0, stream>>>(pc1, W1, b1, b2, b3, wt, out, N);
    softmax_kernel<<<(N + 255) / 256, 256, 0, stream>>>(logits, pc1, out, ind, maxE, minE, N);
    centers_kernel<<<1, 128, 0, stream>>>(maxE, minE, xc);
    flow_kernel<<<(N + 255) / 256, 256, 0, stream>>>(pc1, ind, xc, out, N);
}
